// ContrastiveLossELI5_1838246002814
// MI455X (gfx1250) — compile-verified, matrix operand order checked by reading, not run on hardware
//
#include <hip/hip_runtime.h>
#include <hip/hip_bf16.h>
#include <math.h>

typedef __attribute__((ext_vector_type(16))) _Float16 v16h;
typedef __attribute__((ext_vector_type(8)))  _Float16 v8h;
typedef __attribute__((ext_vector_type(4)))  _Float16 v4h;
typedef __attribute__((ext_vector_type(8)))  float    v8f;

#define NROWS 8192          // 2N
#define HALFN 4096          // N
#define KDIM  256
#define INV_TEMP 2.0f       // 1 / 0.5
#define SMAX 2.0f           // logits are bounded by |cos|/T <= 2 -> fixed lse max

// ---------------------------------------------------------------------------
// Kernel 1: L2-normalize each 256-float row, store as f16 into z[8192][256].
// One wave (32 lanes) per row; 8 floats per lane.
// ---------------------------------------------------------------------------
__global__ __launch_bounds__(256) void normalize_f16_kernel(
    const float* __restrict__ emb_i, const float* __restrict__ emb_j,
    _Float16* __restrict__ z) {
  const int row  = blockIdx.x * 8 + (threadIdx.x >> 5);   // 0..8191
  const int lane = threadIdx.x & 31;
  const float* __restrict__ src =
      (row < HALFN) ? (emb_i + (size_t)row * KDIM)
                    : (emb_j + (size_t)(row - HALFN) * KDIM);

  const float4 a = ((const float4*)src)[lane];
  const float4 b = ((const float4*)src)[lane + 32];

  float ss = a.x*a.x + a.y*a.y + a.z*a.z + a.w*a.w
           + b.x*b.x + b.y*b.y + b.z*b.z + b.w*b.w;
  #pragma unroll
  for (int off = 16; off >= 1; off >>= 1)
    ss += __shfl_xor(ss, off, 32);

  const float inv = 1.0f / fmaxf(sqrtf(ss), 1e-12f);

  _Float16* dst = z + (size_t)row * KDIM;
  v4h lo, hi;
  lo[0] = (_Float16)(a.x*inv); lo[1] = (_Float16)(a.y*inv);
  lo[2] = (_Float16)(a.z*inv); lo[3] = (_Float16)(a.w*inv);
  hi[0] = (_Float16)(b.x*inv); hi[1] = (_Float16)(b.y*inv);
  hi[2] = (_Float16)(b.z*inv); hi[3] = (_Float16)(b.w*inv);
  ((v4h*)dst)[lane]      = lo;
  ((v4h*)dst)[lane + 32] = hi;
}

// ---------------------------------------------------------------------------
// 16-bit 16x32 WMMA operand layout (ISA 7.12.2): lane L (row = L&15) holds
// K = kb..kb+7 in elems 0..7 and K = kb+16..kb+23 in elems 8..15, where
// kb = (L>=16) ? 8 : 0.  z*z^T is symmetric so A and B share this loader.
// ---------------------------------------------------------------------------
__device__ __forceinline__ v16h load_frag(const _Float16* __restrict__ rowp,
                                          int k0, int kb) {
  const v8h lo = *(const v8h*)(rowp + k0 + kb);
  const v8h hi = *(const v8h*)(rowp + k0 + kb + 16);
  v16h f;
  #pragma unroll
  for (int i = 0; i < 8; ++i) { f[i] = lo[i]; f[i + 8] = hi[i]; }
  return f;
}

__device__ __forceinline__ void load_btile(const _Float16* __restrict__ z,
                                           int N0, int lrow, int kb,
                                           v16h* __restrict__ frag) {
  const _Float16* brow = z + (size_t)(N0 + lrow) * KDIM;
  #pragma unroll
  for (int f = 0; f < 8; ++f) frag[f] = load_frag(brow, f * 32, kb);
}

// 8 WMMAs (K=256) + branchless fixed-max softmax accumulation
__device__ __forceinline__ void process_tile(const v16h* __restrict__ a,
                                             const v16h* __restrict__ b,
                                             int M0, int N0, int lrow, int lhalf,
                                             float* __restrict__ runS,
                                             float* __restrict__ posS) {
  v8f c = {};
  #pragma unroll
  for (int f = 0; f < 8; ++f)
    c = __builtin_amdgcn_wmma_f32_16x16x32_f16(
        /*neg_a=*/false, a[f], /*neg_b=*/false, b[f],
        /*c_mod=*/(short)0, c, /*reuse_a=*/false, /*reuse_b=*/false);

  const int gn = N0 + lrow;   // C layout: c[v] = (M = v + 8*lhalf, N = lrow)
  #pragma unroll
  for (int v = 0; v < 8; ++v) {
    const int gm = M0 + v + 8 * lhalf;
    const float s = c[v] * INV_TEMP;
    const float e = __expf(s - SMAX);
    runS[v] += (gn != gm) ? e : 0.0f;                          // mask diagonal
    posS[v] += (gn == ((gm + HALFN) & (NROWS - 1))) ? s : 0.0f; // unique hit
  }
}

// ---------------------------------------------------------------------------
// Kernel 2: each block owns 16 rows. 8 waves sweep 512 column tiles
// (64 tiles/wave) with double-buffered B fragments so VMEM overlaps WMMA.
// rowloss[m] = (SMAX + log(sum_{k!=m} exp(sim[m,k]-SMAX))) - sim[m,(m+N)%2N]
// ---------------------------------------------------------------------------
__global__ __launch_bounds__(256) void simclr_lse_kernel(
    const _Float16* __restrict__ z, float* __restrict__ rowloss) {
  const int M0    = blockIdx.x * 16;
  const int tid   = threadIdx.x;
  const int wave  = tid >> 5;       // 0..7
  const int lane  = tid & 31;
  const int lrow  = lane & 15;
  const int lhalf = lane >> 4;
  const int kb    = lhalf * 8;

  __shared__ float s_sum[8][16];
  __shared__ float s_pos[8][16];

  // A fragments (16 rows x K=256) resident in VGPRs
  v16h afrag[8];
  {
    const _Float16* arow = z + (size_t)(M0 + lrow) * KDIM;
    #pragma unroll
    for (int f = 0; f < 8; ++f) afrag[f] = load_frag(arow, f * 32, kb);
  }

  float runS[8], posS[8];
  #pragma unroll
  for (int v = 0; v < 8; ++v) { runS[v] = 0.0f; posS[v] = 0.0f; }

  // tile j of this wave covers columns N0(j) = wave*16 + 128*j, j = 0..63
  const int nbase = wave * 16;
  v16h b0[8], b1[8];

  load_btile(z, nbase, lrow, kb, b0);
  for (int j = 0; j < 62; j += 2) {
    load_btile(z, nbase + 128 * (j + 1), lrow, kb, b1);      // prefetch
    process_tile(afrag, b0, M0, nbase + 128 * j, lrow, lhalf, runS, posS);
    load_btile(z, nbase + 128 * (j + 2), lrow, kb, b0);      // prefetch
    process_tile(afrag, b1, M0, nbase + 128 * (j + 1), lrow, lhalf, runS, posS);
  }
  load_btile(z, nbase + 128 * 63, lrow, kb, b1);
  process_tile(afrag, b0, M0, nbase + 128 * 62, lrow, lhalf, runS, posS);
  process_tile(afrag, b1, M0, nbase + 128 * 63, lrow, lhalf, runS, posS);

  // reduce across the 16 lanes of each half (same 8 rows per half)
  #pragma unroll
  for (int v = 0; v < 8; ++v) {
    float S = runS[v], P = posS[v];
    #pragma unroll
    for (int off = 1; off < 16; off <<= 1) {
      S += __shfl_xor(S, off, 32);
      P += __shfl_xor(P, off, 32);
    }
    if (lrow == 0) {
      const int m = v + 8 * lhalf;
      s_sum[wave][m] = S;
      s_pos[wave][m] = P;
    }
  }
  __syncthreads();

  // cross-wave reduction + final per-row loss
  if (wave == 0 && lane < 16) {
    float S = 0.0f, P = 0.0f;
    #pragma unroll
    for (int w = 0; w < 8; ++w) { S += s_sum[w][lane]; P += s_pos[w][lane]; }
    rowloss[M0 + lane] = (SMAX + __logf(S)) - P;
  }
}

// ---------------------------------------------------------------------------
// Kernel 3: mean of the 8192 per-row losses -> scalar
// ---------------------------------------------------------------------------
__global__ __launch_bounds__(256) void reduce_mean_kernel(
    const float* __restrict__ rowloss, float* __restrict__ out) {
  __shared__ float sdata[256];
  float acc = 0.0f;
  for (int i = threadIdx.x; i < NROWS; i += 256) acc += rowloss[i];
  sdata[threadIdx.x] = acc;
  __syncthreads();
  #pragma unroll
  for (int s = 128; s > 0; s >>= 1) {
    if (threadIdx.x < s) sdata[threadIdx.x] += sdata[threadIdx.x + s];
    __syncthreads();
  }
  if (threadIdx.x == 0) out[0] = sdata[0] * (1.0f / (float)NROWS);
}

extern "C" void kernel_launch(void* const* d_in, const int* in_sizes, int n_in,
                              void* d_out, int out_size, void* d_ws, size_t ws_size,
                              hipStream_t stream) {
  const float* emb_i = (const float*)d_in[0];
  const float* emb_j = (const float*)d_in[1];
  float* out = (float*)d_out;

  _Float16* z     = (_Float16*)d_ws;                                   // 4 MB
  float*    rowls = (float*)((char*)d_ws + (size_t)NROWS * KDIM * 2);  // 32 KB

  normalize_f16_kernel<<<NROWS / 8, 256, 0, stream>>>(emb_i, emb_j, z);
  simclr_lse_kernel<<<NROWS / 16, 256, 0, stream>>>(z, rowls);
  reduce_mean_kernel<<<1, 256, 0, stream>>>(rowls, out);
}
